// GraphAttention_27960237097148
// MI455X (gfx1250) — compile-verified, matrix operand order checked by reading, not run on hardware
//
#include <hip/hip_runtime.h>
#include <hip/hip_bf16.h>

typedef __bf16 bf16_t;
typedef __attribute__((ext_vector_type(16))) __bf16 v16bf;
typedef __attribute__((ext_vector_type(8)))  __bf16 v8bf;
typedef __attribute__((ext_vector_type(8)))  float  v8f;

#define B_    8
#define CIN_  64
#define N_    512
#define T_    32
#define COUT_ 64
#define TF_   8
#define CT_   25
#define CTP_  32   // c padded to 32 for WMMA K-dim
#define NPW_  8    // n-values processed per workgroup in kernel 1

__device__ __forceinline__ v8f wmma_bf16(v16bf a, v16bf b, v8f c) {
  return __builtin_amdgcn_wmma_f32_16x16x32_bf16(false, a, false, b, (short)0, c, false, false);
}

__device__ __forceinline__ v16bf combine16(v8bf lo, v8bf hi) {
  v16bf r;
  #pragma unroll
  for (int i = 0; i < 8; ++i) { r[i] = lo[i]; r[i + 8] = hi[i]; }
  return r;
}

// ---------------------------------------------------------------------------
// Kernel 0: adjsum = adj + adj_hat   [B,N,N]
// ---------------------------------------------------------------------------
__global__ void adjsum_kernel(const float* __restrict__ a, const float* __restrict__ h,
                              float* __restrict__ s, int n) {
  int i = blockIdx.x * blockDim.x + threadIdx.x;
  if (i < n) s[i] = a[i] + h[i];
}

// ---------------------------------------------------------------------------
// Kernel 1: k/q/v temporal conv as GEMM.  Per (b,n):
//   D[t(32) x c(32pad)] = W[t x z(512)] * Xp[z x c],  z = ci*8+tf.
// 6 waves; wave = (weight wsel, t-tile tt).  All 16 A-fragments of this
// wave's weight rows are cached in REGISTERS (16 x v16bf = 128 VGPRs) and
// reused across NPW_ = 8 n-values -> 8x less weight (L2) traffic.
// Patch staged transposed in LDS ([c][z], stride 520) -> B frags are two
// aligned 16B v8bf loads.  Pad columns handled branchlessly by mask.
// ---------------------------------------------------------------------------
#define XPSTR 520   // halves; 1040B row stride -> lane bank starts 4 apart
__global__ __launch_bounds__(192)
void kqv_conv_kernel(const float* __restrict__ x,
                     const float* __restrict__ Wk, const float* __restrict__ bk,
                     const float* __restrict__ Wq, const float* __restrict__ bq,
                     const float* __restrict__ Wv, const float* __restrict__ bv,
                     bf16_t* __restrict__ Kb, bf16_t* __restrict__ Qb, bf16_t* __restrict__ Vb)
{
  __shared__ __align__(16) bf16_t xl[CIN_ * T_];        // raw x row [ci][t]  (4 KB)
  __shared__ __align__(16) bf16_t sXpT[CTP_ * XPSTR];   // patch [c][z]       (33 KB)
  const int b    = blockIdx.x / (N_ / NPW_);
  const int ngrp = blockIdx.x % (N_ / NPW_);
  const int tid  = threadIdx.x;
  const int lane = tid & 31;
  const int w    = tid >> 5;                 // 0..5
  const int half = lane >> 4;
  const int lr   = lane & 15;
  const int wsel = w % 3;                    // which weight (k/q/v)
  const int t0   = (w / 3) * 16;             // t-tile

  const float* Wsel = (wsel == 0) ? Wk : (wsel == 1) ? Wq : Wv;
  const float* bsel = (wsel == 0) ? bk : (wsel == 1) ? bq : bv;
  bf16_t*      Osel = (wsel == 0) ? Kb : (wsel == 1) ? Qb : Vb;

  // preload A fragments for this wave's 16 weight rows, all K (once per WG)
  v16bf akf[16];
  #pragma unroll
  for (int kk = 0; kk < 16; ++kk) {
    const int wrow = (t0 + lr) * 512 + kk * 32;
    #pragma unroll
    for (int i = 0; i < 8; ++i) {
      akf[kk][i]     = (bf16_t)Wsel[wrow + half * 8 + i];
      akf[kk][i + 8] = (bf16_t)Wsel[wrow + 16 + half * 8 + i];
    }
  }
  // preload bias for this wave's 8 output rows
  float bias8[8];
  #pragma unroll
  for (int r = 0; r < 8; ++r) bias8[r] = bsel[t0 + half * 8 + r];

  const float mask1 = (lr < CT_ - 16) ? 1.0f : 0.0f;   // cols 16..31 validity
  const v8bf* xp0 = (const v8bf*)(sXpT + lr * XPSTR);          // cols 0..15
  const v8bf* xp1 = (const v8bf*)(sXpT + (16 + lr) * XPSTR);   // cols 16..31

  for (int ni = 0; ni < NPW_; ++ni) {
    const int n = ngrp * NPW_ + ni;
    __syncthreads();                          // previous iter's reads done

    // stage x[b,:,n,:] (64x32) into LDS as bf16
    for (int idx = tid; idx < CIN_ * T_; idx += 192) {
      int ci = idx >> 5, tt2 = idx & 31;
      xl[idx] = (bf16_t)x[((size_t)(b * CIN_ + ci) * N_ + n) * T_ + tt2];
    }
    __syncthreads();

    // build transposed im2col patch: sXpT[c][z] = x[ci][min(c+tf,31)]
    for (int idx = tid; idx < CTP_ * 512; idx += 192) {
      int cc = idx >> 9, z = idx & 511;
      int tt2 = cc + (z & 7);
      tt2 = (tt2 > 31) ? 31 : tt2;            // pad cols: value irrelevant
      sXpT[cc * XPSTR + z] = xl[((z >> 3) << 5) + tt2];
    }
    __syncthreads();

    v8f acc0 = {0,0,0,0,0,0,0,0};
    v8f acc1 = {0,0,0,0,0,0,0,0};
    #pragma unroll
    for (int kk = 0; kk < 16; ++kk) {
      v16bf bx0 = combine16(xp0[kk * 4 + half], xp0[kk * 4 + 2 + half]);
      v16bf bx1 = combine16(xp1[kk * 4 + half], xp1[kk * 4 + 2 + half]);
      acc0 = wmma_bf16(akf[kk], bx0, acc0);
      acc1 = wmma_bf16(akf[kk], bx1, acc1);
    }

    // branchless store (+bias); pad columns (>=25) multiplied to exactly 0
    #pragma unroll
    for (int r = 0; r < 8; ++r) {
      int t = t0 + half * 8 + r;              // D row mapping
      size_t o = ((size_t)(b * T_ + t) * N_ + n) * CTP_;
      Osel[o + lr]      = (bf16_t)(acc0[r] + bias8[r]);            // cols 0..15 always valid
      Osel[o + 16 + lr] = (bf16_t)((acc1[r] + bias8[r]) * mask1);  // cols 16..31 masked
    }
  }
}

// ---------------------------------------------------------------------------
// Kernel 2: fused attention per (b,t), flash-style online softmax.
// WG = 8 waves = 128 rows of n; wave owns 16 rows. 16 steps of 32 m-columns:
//   S(16x32) via 2 WMMA (K=c=32), +adjsum, online softmax,
//   P relaid D->A through per-wave bf16 LDS tile, O += P*V via 2 WMMA (K=m=32).
// ---------------------------------------------------------------------------
#define VSTR 520   // sVt row stride in halves
#define PSTR 40    // sP row stride in halves (80B, 16B-aligned rows)
__global__ __launch_bounds__(256)
void attn_kernel(const bf16_t* __restrict__ Kb, const bf16_t* __restrict__ Qb,
                 const bf16_t* __restrict__ Vb, const float* __restrict__ adjs,
                 float* __restrict__ Ob)
{
  __shared__ __align__(16) bf16_t sVt[CTP_ * VSTR];     // V transposed [c][m] (33.3 KB)
  __shared__ __align__(16) bf16_t sPb[8 * 16 * PSTR];   // per-wave P staging  (10.2 KB)

  const int nb  = blockIdx.x & 3;
  const int t   = (blockIdx.x >> 2) & 31;
  const int b   = blockIdx.x >> 7;
  const int tid = threadIdx.x;
  const size_t base = (size_t)(b * T_ + t) * N_ * CTP_;

  // transpose V[b,t] (512x32) into LDS [c][m]
  for (int j = 0; j < 64; ++j) {
    int idx = tid + j * 256;
    int m = idx >> 5, cc = idx & 31;
    sVt[cc * VSTR + m] = Vb[base + idx];
  }
  __syncthreads();

  const int lane = tid & 31;
  const int w    = tid >> 5;
  const int half = lane >> 4;
  const int lr   = lane & 15;
  const int n0   = nb * 128 + w * 16;

  const v8bf* kp = (const v8bf*)(Kb + base + (size_t)(n0 + lr) * CTP_);
  v16bf ka = combine16(kp[half], kp[2 + half]);

  float m_run[8], l_run[8];
  #pragma unroll
  for (int r = 0; r < 8; ++r) { m_run[r] = -3.0e38f; l_run[r] = 0.0f; }
  v8f o0 = {0,0,0,0,0,0,0,0};
  v8f o1 = {0,0,0,0,0,0,0,0};

  const size_t adjbase = (size_t)b * N_ * N_;
  const v8bf* vrow0 = (const v8bf*)(sVt + lr * VSTR);
  const v8bf* vrow1 = (const v8bf*)(sVt + (16 + lr) * VSTR);
  const v8bf* prow  = (const v8bf*)(sPb + (w * 16 + lr) * PSTR);
  bf16_t* pw = sPb + w * 16 * PSTR;

  for (int mb = 0; mb < 16; ++mb) {
    const v8bf* qp0 = (const v8bf*)(Qb + base + (size_t)(mb * 32 + lr) * CTP_);
    const v8bf* qp1 = (const v8bf*)(Qb + base + (size_t)(mb * 32 + 16 + lr) * CTP_);
    v16bf q0 = combine16(qp0[half], qp0[2 + half]);
    v16bf q1 = combine16(qp1[half], qp1[2 + half]);

    v8f zero = {0,0,0,0,0,0,0,0};
    v8f s0 = wmma_bf16(ka, q0, zero);
    v8f s1 = wmma_bf16(ka, q1, zero);

    #pragma unroll
    for (int r = 0; r < 8; ++r) {
      int row = n0 + half * 8 + r;
      size_t ao = adjbase + (size_t)row * N_ + mb * 32 + lr;
      s0[r] += adjs[ao];
      s1[r] += adjs[ao + 16];
      if (mb < 15) __builtin_prefetch(&adjs[ao + 32], 0, 1);
    }

    float p0[8], p1[8];
    #pragma unroll
    for (int r = 0; r < 8; ++r) {
      float mx = fmaxf(s0[r], s1[r]);
      mx = fmaxf(mx, __shfl_xor(mx, 1));
      mx = fmaxf(mx, __shfl_xor(mx, 2));
      mx = fmaxf(mx, __shfl_xor(mx, 4));
      mx = fmaxf(mx, __shfl_xor(mx, 8));
      float mnew = fmaxf(m_run[r], mx);
      float sc   = __expf(m_run[r] - mnew);
      float e0   = __expf(s0[r] - mnew);
      float e1   = __expf(s1[r] - mnew);
      float rs = e0 + e1;
      rs += __shfl_xor(rs, 1);
      rs += __shfl_xor(rs, 2);
      rs += __shfl_xor(rs, 4);
      rs += __shfl_xor(rs, 8);
      l_run[r] = l_run[r] * sc + rs;
      m_run[r] = mnew;
      p0[r] = e0;  p1[r] = e1;
      o0[r] *= sc; o1[r] *= sc;
    }

    #pragma unroll
    for (int r = 0; r < 8; ++r) {
      pw[(half * 8 + r) * PSTR + lr]      = (bf16_t)p0[r];
      pw[(half * 8 + r) * PSTR + 16 + lr] = (bf16_t)p1[r];
    }
    asm volatile("s_wait_dscnt 0" ::: "memory");
    v16bf pa = combine16(prow[half], prow[2 + half]);

    v16bf vb0 = combine16(vrow0[mb * 4 + half], vrow0[mb * 4 + 2 + half]);
    v16bf vb1 = combine16(vrow1[mb * 4 + half], vrow1[mb * 4 + 2 + half]);

    o0 = wmma_bf16(pa, vb0, o0);
    o1 = wmma_bf16(pa, vb1, o1);
    asm volatile("s_wait_dscnt 0" ::: "memory");
  }

  #pragma unroll
  for (int r = 0; r < 8; ++r) {
    float inv = 1.0f / l_run[r];
    int row = n0 + half * 8 + r;
    size_t o = base + (size_t)row * CTP_;
    Ob[o + lr]      = o0[r] * inv;
    Ob[o + 16 + lr] = o1[r] * inv;
  }
}

// ---------------------------------------------------------------------------
// Kernel 3a: Y = Wout*O + bout + Wres*x + bres ; accumulate BN sums per channel
// ---------------------------------------------------------------------------
__global__ __launch_bounds__(256)
void fuse_kernel(const float* __restrict__ x, const float* __restrict__ Ob,
                 const float* __restrict__ Wres, const float* __restrict__ bres,
                 const float* __restrict__ Wout, const float* __restrict__ bout,
                 float* __restrict__ Y, float* __restrict__ gstats)
{
  __shared__ float sWr[COUT_ * CIN_];
  __shared__ float sWo[COUT_ * CT_];
  __shared__ float sBias[COUT_];
  __shared__ float ssum[COUT_], ssq[COUT_];
  const int tid = threadIdx.x;
  for (int i = tid; i < COUT_ * CIN_; i += 256) sWr[i] = Wres[i];
  for (int i = tid; i < COUT_ * CT_;  i += 256) sWo[i] = Wout[i];
  if (tid < COUT_) { sBias[tid] = bres[tid] + bout[tid]; ssum[tid] = 0.0f; ssq[tid] = 0.0f; }
  __syncthreads();

  const int b = blockIdx.x >> 6;
  const int n = (blockIdx.x & 63) * 8 + (tid >> 5);
  const int t = tid & 31;

  float xv[CIN_];
  #pragma unroll
  for (int ci = 0; ci < CIN_; ++ci)
    xv[ci] = x[((size_t)(b * CIN_ + ci) * N_ + n) * T_ + t];
  float ov[CT_];
  #pragma unroll
  for (int cc = 0; cc < CT_; ++cc)
    ov[cc] = Ob[((size_t)(b * T_ + t) * N_ + n) * CTP_ + cc];

  for (int co = 0; co < COUT_; ++co) {
    float acc = sBias[co];
    #pragma unroll
    for (int ci = 0; ci < CIN_; ++ci) acc = fmaf(sWr[co * CIN_ + ci], xv[ci], acc);
    #pragma unroll
    for (int cc = 0; cc < CT_; ++cc)  acc = fmaf(sWo[co * CT_ + cc], ov[cc], acc);
    Y[((size_t)(b * COUT_ + co) * N_ + n) * T_ + t] = acc;
    float s = acc, q = acc * acc;
    for (int msk = 16; msk >= 1; msk >>= 1) { s += __shfl_xor(s, msk); q += __shfl_xor(q, msk); }
    if ((tid & 31) == 0) { atomicAdd(&ssum[co], s); atomicAdd(&ssq[co], q); }
  }
  __syncthreads();
  if (tid < COUT_) {
    atomicAdd(&gstats[tid], ssum[tid]);
    atomicAdd(&gstats[COUT_ + tid], ssq[tid]);
  }
}

// ---------------------------------------------------------------------------
// Kernel 3b: batchnorm normalize into d_out
// ---------------------------------------------------------------------------
__global__ void bn_kernel(const float* __restrict__ Y, const float* __restrict__ gstats,
                          const float* __restrict__ gamma, const float* __restrict__ beta,
                          float* __restrict__ out)
{
  int idx = blockIdx.x * 256 + threadIdx.x;
  int co = (idx >> 14) & (COUT_ - 1);            // N_*T_ = 16384
  const float Mi = 1.0f / (float)(B_ * N_ * T_);
  float mean = gstats[co] * Mi;
  float var  = gstats[COUT_ + co] * Mi - mean * mean;
  out[idx] = (Y[idx] - mean) * rsqrtf(var + 1e-5f) * gamma[co] + beta[co];
}

// ---------------------------------------------------------------------------
extern "C" void kernel_launch(void* const* d_in, const int* in_sizes, int n_in,
                              void* d_out, int out_size, void* d_ws, size_t ws_size,
                              hipStream_t stream)
{
  const float* x    = (const float*)d_in[0];
  const float* adj  = (const float*)d_in[1];
  const float* adjh = (const float*)d_in[2];
  const float* Wk   = (const float*)d_in[3];  const float* bk   = (const float*)d_in[4];
  const float* Wq   = (const float*)d_in[5];  const float* bq   = (const float*)d_in[6];
  const float* Wv   = (const float*)d_in[7];  const float* bv   = (const float*)d_in[8];
  const float* Wres = (const float*)d_in[9];  const float* bres = (const float*)d_in[10];
  const float* Wout = (const float*)d_in[11]; const float* bout = (const float*)d_in[12];
  const float* gamma = (const float*)d_in[13]; const float* beta = (const float*)d_in[14];
  float* out = (float*)d_out;

  char* ws = (char*)d_ws;
  const size_t kqv_elems = (size_t)B_ * T_ * N_ * CTP_;     // 4,194,304
  bf16_t* Kb   = (bf16_t*)ws;  ws += kqv_elems * sizeof(bf16_t);
  bf16_t* Qb   = (bf16_t*)ws;  ws += kqv_elems * sizeof(bf16_t);
  bf16_t* Vb   = (bf16_t*)ws;  ws += kqv_elems * sizeof(bf16_t);
  float*  adjs = (float*)ws;   ws += (size_t)B_ * N_ * N_ * sizeof(float);
  float*  Ob   = (float*)ws;   ws += kqv_elems * sizeof(float);
  float*  Y    = (float*)ws;   ws += (size_t)B_ * COUT_ * N_ * T_ * sizeof(float);
  float*  gstats = (float*)ws;

  hipMemsetAsync(gstats, 0, 2 * COUT_ * sizeof(float), stream);

  adjsum_kernel<<<(B_ * N_ * N_) / 256, 256, 0, stream>>>(adj, adjh, adjs, B_ * N_ * N_);
  kqv_conv_kernel<<<B_ * (N_ / NPW_), 192, 0, stream>>>(x, Wk, bk, Wq, bq, Wv, bv, Kb, Qb, Vb);
  attn_kernel<<<B_ * T_ * 4, 256, 0, stream>>>(Kb, Qb, Vb, adjs, Ob);
  fuse_kernel<<<B_ * (N_ / 8), 256, 0, stream>>>(x, Ob, Wres, bres, Wout, bout, Y, gstats);
  bn_kernel<<<(B_ * COUT_ * N_ * T_) / 256, 256, 0, stream>>>(Y, gstats, gamma, beta, out);
}
